// SeedPool_56599079027413
// MI455X (gfx1250) — matrix operand order repaired and checked by reading, not compiled, not run on hardware
//
#include <hip/hip_runtime.h>
#include <hip/hip_bf16.h>
#include <stdint.h>

typedef unsigned int       u32;
typedef unsigned short     u16;
typedef unsigned long long u64;
typedef __bf16 bf16;
typedef bf16  v16bf __attribute__((ext_vector_type(16)));
typedef float v8f   __attribute__((ext_vector_type(8)));

// exact parameter types of the async-to-LDS builtins (per compiler diagnostic)
typedef int v4i_ __attribute__((vector_size(16)));
typedef int v2i_ __attribute__((vector_size(8)));

#define AS1 __attribute__((address_space(1)))
#define AS3 __attribute__((address_space(3)))

union AccF { v8f v; float f[8]; };
union FragBF { v16bf v; u32 u[8]; };

// native f32 -> bf16 (backend emits hardware cvt on gfx1250)
__device__ __forceinline__ u16 f2bf(float f) { return __builtin_bit_cast(u16, (bf16)f); }

// ---------------------------------------------------------------------------
// CDNA5 async global->LDS copy (ASYNCcnt-tracked DMA).
// ---------------------------------------------------------------------------
#if __has_builtin(__builtin_amdgcn_s_wait_asynccnt)
#define WAITA(n) __builtin_amdgcn_s_wait_asynccnt(n)
#else
#define WAITA(n) asm volatile("s_wait_asynccnt " #n ::: "memory")
#endif

__device__ __forceinline__ void async_copy_b128(const float* g, float* l) {
#if __has_builtin(__builtin_amdgcn_global_load_async_to_lds_b128)
    __builtin_amdgcn_global_load_async_to_lds_b128((AS1 v4i_*)g, (AS3 v4i_*)l, 0, 0);
#else
    u32 loff = (u32)(uintptr_t)(AS3 void*)l;
    asm volatile("global_load_async_to_lds_b128 %0, %1, off"
                 :: "v"(loff), "v"((u64)(uintptr_t)g) : "memory");
#endif
}

__device__ __forceinline__ void async_copy_b64(const u16* g, u16* l) {
#if __has_builtin(__builtin_amdgcn_global_load_async_to_lds_b64)
    __builtin_amdgcn_global_load_async_to_lds_b64((AS1 v2i_*)g, (AS3 v2i_*)l, 0, 0);
#else
    u32 loff = (u32)(uintptr_t)(AS3 void*)l;
    asm volatile("global_load_async_to_lds_b64 %0, %1, off"
                 :: "v"(loff), "v"((u64)(uintptr_t)g) : "memory");
#endif
}

// ---------------------------------------------------------------------------
// Fragment builders (convert f32 LDS tiles to bf16 at build time).
// A (16x32): lanes 0-15 row=lane, K={0..7,16..23}; lanes 16-31 K={8..15,24..31}.
// B (32x16): lanes 0-15 N=lane K={0..7,16..23}; lanes 16-31 K={8..15,24..31}.
// ---------------------------------------------------------------------------
__device__ __forceinline__ v16bf load_a_frag_f32(const float* lds, int stride,
                                                 int row_base, int lane) {
    const int row   = row_base + (lane & 15);
    const int khalf = (lane >> 4) << 3;
    const float* rp = lds + row * stride + khalf;
    v16bf v;
#pragma unroll
    for (int j = 0; j < 8; ++j) { v[j] = (bf16)rp[j]; v[8 + j] = (bf16)rp[16 + j]; }
    return v;
}

__device__ __forceinline__ v16bf load_b_frag_f32(const float* lds, int stride,
                                                 int col_base, int lane) {
    const int n    = col_base + (lane & 15);
    const int koff = (lane >> 4) << 3;
    const float* rp = lds + n * stride + koff;
    v16bf v;
#pragma unroll
    for (int j = 0; j < 8; ++j) { v[j] = (bf16)rp[j]; v[8 + j] = (bf16)rp[16 + j]; }
    return v;
}

// A fragment from a bf16(u16) LDS tile (attn is already bf16)
__device__ __forceinline__ v16bf load_a_frag_u16(const u16* lds, int stride,
                                                 int row_base, int lane) {
    const int row = row_base + (lane & 15);
    const int o0  = ((lane >> 4) << 3) >> 1;
    const u32* rp = (const u32*)(lds + row * stride);
    FragBF f;
#pragma unroll
    for (int j = 0; j < 4; ++j) { f.u[j] = rp[o0 + j]; f.u[4 + j] = rp[8 + o0 + j]; }
    return f.v;
}

// B fragment gathered from [k][d] f32 tile (column read, conflict-free per half-wave)
__device__ __forceinline__ v16bf load_b_frag_gather(const float* lds, int stride,
                                                    int col_base, int lane) {
    const int d    = col_base + (lane & 15);
    const int koff = (lane >> 4) << 3;
    const float* p = lds + koff * stride + d;
    v16bf v;
#pragma unroll
    for (int j = 0; j < 8; ++j) { v[j] = (bf16)p[j * stride]; v[8 + j] = (bf16)p[(16 + j) * stride]; }
    return v;
}

__device__ __forceinline__ v8f wmma_bf16(v16bf a, v16bf b, v8f c) {
    return __builtin_amdgcn_wmma_f32_16x16x32_bf16(false, a, false, b,
                                                   (short)0, c, false, false);
}

// ---------------------------------------------------------------------------
// Common 32xK x 128 GEMM with async double-buffered staging.
// A: 32 rows (stride aStride, K contiguous). B: 128 N-rows (stride bStride).
// sa: 2 x 32x36 floats, sb: 2 x 128x36 floats (padded rows, 16B aligned).
// ---------------------------------------------------------------------------
__device__ __forceinline__ void stage_issue_32x128(const float* aBase, size_t aStride,
                                                   const float* bBase, size_t bStride,
                                                   int k0, float* sa, float* sb, int tid) {
    {   // A tile: 32x32 f32 = 4KB, one b128 per thread
        const int s = tid >> 3, c = (tid & 7) * 4;
        async_copy_b128(aBase + (size_t)s * aStride + k0 + c, sa + s * 36 + c);
    }
#pragma unroll
    for (int j = 0; j < 4; ++j) {   // B tile: 128x32 f32 = 16KB, 4 b128 per thread
        const int id = tid + 256 * j;
        const int n = id >> 3, c = (id & 7) * 4;
        async_copy_b128(bBase + (size_t)n * bStride + k0 + c, sb + n * 36 + c);
    }
}

__device__ __forceinline__ void gemm_32xK_128_async(const float* aBase, size_t aStride,
                                                    const float* bBase, size_t bStride,
                                                    float* outBase, int outStride,
                                                    float scale, int ksteps,
                                                    float* sa, float* sb) {
    const int tid = threadIdx.x, lane = tid & 31, wave = tid >> 5;
    v8f acc0 = {}, acc1 = {};
    stage_issue_32x128(aBase, aStride, bBase, bStride, 0, sa, sb, tid);
    for (int i = 0; i < ksteps; ++i) {
        const int cur = i & 1;
        __syncthreads();                       // prior readers of next buffer done
        if (i + 1 < ksteps) {
            stage_issue_32x128(aBase, aStride, bBase, bStride, (i + 1) * 32,
                               sa + (1 - cur) * 32 * 36, sb + (1 - cur) * 128 * 36, tid);
            WAITA(5);                          // buffer i complete; i+1 still in flight
        } else {
            WAITA(0);
        }
        __syncthreads();                       // buffer i visible to all waves
        const float* A = sa + cur * 32 * 36;
        const float* B = sb + cur * 128 * 36;
        v16bf a0 = load_a_frag_f32(A, 36, 0,  lane);
        v16bf a1 = load_a_frag_f32(A, 36, 16, lane);
        v16bf bb = load_b_frag_f32(B, 36, wave * 16, lane);
        acc0 = wmma_bf16(a0, bb, acc0);
        acc1 = wmma_bf16(a1, bb, acc1);
    }
    const int col  = wave * 16 + (lane & 15);
    const int mofs = (lane >> 4) * 8;
    AccF u0, u1; u0.v = acc0; u1.v = acc1;
#pragma unroll
    for (int r = 0; r < 8; ++r) {
        outBase[(size_t)(mofs + r)      * outStride + col] = u0.f[r] * scale;
        outBase[(size_t)(16 + mofs + r) * outStride + col] = u1.f[r] * scale;
    }
}

// ---------------------------------------------------------------------------
// K1: scores[b][s][n] = seeds @ x^T / sqrt(D).  Grid: 32 batches x 32 n-tiles.
// ---------------------------------------------------------------------------
__global__ __launch_bounds__(256)
void k_scores(const float* __restrict__ x, const float* __restrict__ seeds,
              float* __restrict__ scores) {
    __shared__ float sa[2 * 32 * 36];
    __shared__ float sb[2 * 128 * 36];
    const int b = blockIdx.x >> 5, nt = blockIdx.x & 31;
    gemm_32xK_128_async(seeds, 1024,
                        x + (size_t)b * 4096 * 1024 + (size_t)nt * 128 * 1024, 1024,
                        scores + (size_t)b * 32 * 4096 + nt * 128, 4096,
                        0.03125f /*1/sqrt(1024)*/, 32, sa, sb);
}

// ---------------------------------------------------------------------------
// K2: softmax over N=4096 per (b,s) row; write attn as bf16.
// ---------------------------------------------------------------------------
__global__ __launch_bounds__(256)
void k_softmax(const float* __restrict__ scores, u16* __restrict__ attn) {
    __shared__ float red[256];
    const int row = blockIdx.x;
    const float* sp = scores + (size_t)row * 4096;
    u16* ap = attn + (size_t)row * 4096;
    const int tid = threadIdx.x;

    float m = -1e30f;
    for (int i = tid; i < 4096; i += 256) m = fmaxf(m, sp[i]);
    red[tid] = m; __syncthreads();
    for (int s = 128; s > 0; s >>= 1) {
        if (tid < s) red[tid] = fmaxf(red[tid], red[tid + s]);
        __syncthreads();
    }
    m = red[0]; __syncthreads();

    float sum = 0.f;
    for (int i = tid; i < 4096; i += 256) sum += __expf(sp[i] - m);
    red[tid] = sum; __syncthreads();
    for (int s = 128; s > 0; s >>= 1) {
        if (tid < s) red[tid] += red[tid + s];
        __syncthreads();
    }
    const float inv = 1.0f / red[0];
    for (int i = tid; i < 4096; i += 256) ap[i] = f2bf(__expf(sp[i] - m) * inv);
}

// ---------------------------------------------------------------------------
// K3: pooled[b][s][d] = attn @ x.  x tile async-staged in native [k][d] layout;
// B fragment gathered column-wise.  Grid: 32 batches x 8 d-tiles.
// ---------------------------------------------------------------------------
__global__ __launch_bounds__(256)
void k_pool(const u16* __restrict__ attn, const float* __restrict__ x,
            float* __restrict__ pooled) {
    __shared__ u16   sa[2 * 32 * 36];      // attn tile [s][k], bf16
    __shared__ float sb[2 * 32 * 132];     // x tile [k][d], f32, padded
    const int b = blockIdx.x >> 3, dt = blockIdx.x & 7;
    const int tid = threadIdx.x, lane = tid & 31, wave = tid >> 5;
    const u16*   ab = attn + (size_t)b * 32 * 4096;
    const float* xb = x + (size_t)b * 4096 * 1024 + dt * 128;

    auto issue = [&](int buf, int k0) {
        {   // attn: 32x32 bf16 = 2KB, one b64 per thread
            const int s = tid >> 3, c = (tid & 7) * 4;
            async_copy_b64(ab + (size_t)s * 4096 + k0 + c, sa + buf * 32 * 36 + s * 36 + c);
        }
#pragma unroll
        for (int j = 0; j < 4; ++j) {   // x: 32x128 f32 = 16KB, 4 b128 per thread
            const int id = tid + 256 * j;
            const int r = id >> 5, c = (id & 31) * 4;
            async_copy_b128(xb + (size_t)(k0 + r) * 1024 + c, sb + buf * 32 * 132 + r * 132 + c);
        }
    };

    v8f acc0 = {}, acc1 = {};
    issue(0, 0);
    for (int i = 0; i < 128; ++i) {
        const int cur = i & 1;
        __syncthreads();
        if (i + 1 < 128) { issue(1 - cur, (i + 1) * 32); WAITA(5); }
        else             { WAITA(0); }
        __syncthreads();
        const u16*   A = sa + cur * 32 * 36;
        const float* B = sb + cur * 32 * 132;
        v16bf a0 = load_a_frag_u16(A, 36, 0,  lane);
        v16bf a1 = load_a_frag_u16(A, 36, 16, lane);
        v16bf bb = load_b_frag_gather(B, 132, wave * 16, lane);
        acc0 = wmma_bf16(a0, bb, acc0);
        acc1 = wmma_bf16(a1, bb, acc1);
    }
    const int dglob = dt * 128 + wave * 16 + (lane & 15);
    const int mofs  = (lane >> 4) * 8;
    float* pb = pooled + (size_t)b * 32 * 1024;
    AccF u0, u1; u0.v = acc0; u1.v = acc1;
#pragma unroll
    for (int r = 0; r < 8; ++r) {
        pb[(size_t)(mofs + r)      * 1024 + dglob] = u0.f[r];
        pb[(size_t)(16 + mofs + r) * 1024 + dglob] = u1.f[r];
    }
}

// ---------------------------------------------------------------------------
// K4: projection partials with K-split.  Grid: 8 o-tiles x 32 k-splits.
// ---------------------------------------------------------------------------
__global__ __launch_bounds__(256)
void k_proj(const float* __restrict__ pooled, const float* __restrict__ W,
            float* __restrict__ part) {
    __shared__ float sa[2 * 32 * 36];
    __shared__ float sb[2 * 128 * 36];
    const int ot = blockIdx.x >> 5, ks = blockIdx.x & 31;
    const size_t kbase = (size_t)ks * 1024;
    gemm_32xK_128_async(pooled + kbase, 32768,
                        W + (size_t)ot * 128 * 32768 + kbase, 32768,
                        part + (size_t)ks * 32 * 1024 + ot * 128, 1024,
                        1.0f, 32, sa, sb);
}

// ---------------------------------------------------------------------------
// K5: out[b][o] = bias[o] + sum_ks part[ks][b][o]
// ---------------------------------------------------------------------------
__global__ __launch_bounds__(256)
void k_reduce(const float* __restrict__ part, const float* __restrict__ bias,
              float* __restrict__ out) {
    const int i = blockIdx.x * 256 + threadIdx.x;
    const int o = i & 1023;
    float s = bias[o];
#pragma unroll 4
    for (int ks = 0; ks < 32; ++ks) s += part[(size_t)ks * 32768 + i];
    out[i] = s;
}

extern "C" void kernel_launch(void* const* d_in, const int* in_sizes, int n_in,
                              void* d_out, int out_size, void* d_ws, size_t ws_size,
                              hipStream_t stream) {
    (void)in_sizes; (void)n_in; (void)out_size; (void)ws_size;
    const float* x     = (const float*)d_in[0];   // (32,4096,1024)
    const float* seeds = (const float*)d_in[1];   // (32,1024)
    const float* W     = (const float*)d_in[2];   // (1024,32768)
    const float* bias  = (const float*)d_in[3];   // (1024)
    float* out = (float*)d_out;                   // (32,1024)

    float* ws_scores = (float*)d_ws;                                  // 16 MB
    u16*   ws_attn   = (u16*)((char*)d_ws + (16u << 20));             //  8 MB
    float* ws_pooled = (float*)((char*)d_ws + (24u << 20));           //  4 MB
    float* ws_part   = (float*)((char*)d_ws + (28u << 20));           //  4 MB

    k_scores <<<1024, 256, 0, stream>>>(x, seeds, ws_scores);
    k_softmax<<<1024, 256, 0, stream>>>(ws_scores, ws_attn);
    k_pool   <<<256,  256, 0, stream>>>(ws_attn, x, ws_pooled);
    k_proj   <<<256,  256, 0, stream>>>(ws_pooled, W, ws_part);
    k_reduce <<<128,  256, 0, stream>>>(ws_part, bias, out);
}
